// Quantizer_65420941853382
// MI455X (gfx1250) — compile-verified, matrix operand order checked by reading, not run on hardware
//
#include <hip/hip_runtime.h>
#include <hip/hip_bf16.h>
#include <math.h>

typedef float    v2f  __attribute__((ext_vector_type(2)));
typedef float    v8f  __attribute__((ext_vector_type(8)));
typedef _Float16 v8h  __attribute__((ext_vector_type(8)));
typedef _Float16 v16h __attribute__((ext_vector_type(16)));

#define N_TOKENS   32768
#define D_MODEL    1024
#define NUM_CODES  4096
#define CODE_DEPTH 256

// ---------------------------------------------------------------------------
// Kernel 0: per-code squared norm (code 0 -> +inf so it is never selected)
//           + codebook converted to f16 for the WMMA distance GEMM.
// One wave per codebook row.
// ---------------------------------------------------------------------------
__global__ __launch_bounds__(256)
void prep_kernel(const float* __restrict__ cb,
                 float* __restrict__ cnorm,
                 _Float16* __restrict__ cb16) {
    const int gwave = (blockIdx.x * 256 + threadIdx.x) >> 5;
    const int lane  = threadIdx.x & 31;
    if (gwave >= NUM_CODES) return;
    const float* row = cb + (size_t)gwave * CODE_DEPTH;
    float s = 0.0f;
#pragma unroll
    for (int j = 0; j < 8; ++j) {
        float v = row[lane * 8 + j];
        s += v * v;
        cb16[(size_t)gwave * CODE_DEPTH + lane * 8 + j] = (_Float16)v;
    }
#pragma unroll
    for (int m = 1; m < 32; m <<= 1) s += __shfl_xor(s, m, 32);
    if (lane == 0) cnorm[gwave] = (gwave == 0) ? 1e30f : s;
}

// ---------------------------------------------------------------------------
// Kernel 1: z = normalize(x @ W_down + b_down), f32 WMMA (16x16x4).
// 512 threads = 16 waves; wave w owns column tile [16w, 16w+16).
// K staged through LDS in chunks of 32. f32 WMMA keeps the returned z exact.
// ---------------------------------------------------------------------------
__global__ __launch_bounds__(512)
void down_kernel(const float* __restrict__ x,
                 const float* __restrict__ W,
                 const float* __restrict__ bias,
                 float* __restrict__ zout) {
    __shared__ float xs[16][32];        //  2 KB
    __shared__ float ws[32][256];       // 32 KB
    __shared__ float zb[16][256];       // 16 KB

    const int tid     = threadIdx.x;
    const int wave    = tid >> 5;
    const int lane    = tid & 31;
    const int rowBase = blockIdx.x * 16;
    const int colBase = wave * 16;
    const int m = lane & 15;            // M for A, N for B/C
    const int h = lane >> 4;

    v8f acc = {};
    for (int kBase = 0; kBase < D_MODEL; kBase += 32) {
        // stage x tile: 16x32 = 512 floats, one per thread
        {
            const int r = tid >> 5, k = tid & 31;
            xs[r][k] = x[(size_t)(rowBase + r) * D_MODEL + kBase + k];
        }
        // stage W tile: 32x256 = 2048 float4, four per thread
#pragma unroll
        for (int j = 0; j < 4; ++j) {
            const int o  = (tid + 512 * j) * 4;
            const int kr = o >> 8, c = o & 255;
            *(float4*)&ws[kr][c] =
                *(const float4*)&W[(size_t)(kBase + kr) * CODE_DEPTH + c];
        }
        __syncthreads();
#pragma unroll
        for (int kk = 0; kk < 32; kk += 4) {
            v2f a, b;
            a.x = xs[m][kk + 2 * h];
            a.y = xs[m][kk + 2 * h + 1];
            b.x = ws[kk + 2 * h][colBase + m];
            b.y = ws[kk + 2 * h + 1][colBase + m];
            acc = __builtin_amdgcn_wmma_f32_16x16x4_f32(
                false, a, false, b, (short)0, acc, false, false);
        }
        __syncthreads();
    }

    // bias + spill the 16x256 strip to LDS for row-normalization
    const float bb = bias[colBase + m];
#pragma unroll
    for (int v = 0; v < 8; ++v) zb[v + 8 * h][colBase + m] = acc[v] + bb;
    __syncthreads();

    // wave w normalizes row w: 256 elems = 32 lanes * 8
    float4 p0 = *(float4*)&zb[wave][lane * 8];
    float4 p1 = *(float4*)&zb[wave][lane * 8 + 4];
    float s = p0.x * p0.x + p0.y * p0.y + p0.z * p0.z + p0.w * p0.w
            + p1.x * p1.x + p1.y * p1.y + p1.z * p1.z + p1.w * p1.w;
#pragma unroll
    for (int mm = 1; mm < 32; mm <<= 1) s += __shfl_xor(s, mm, 32);
    const float inv = 1.0f / (sqrtf(s) + 1e-6f);
    p0.x *= inv; p0.y *= inv; p0.z *= inv; p0.w *= inv;
    p1.x *= inv; p1.y *= inv; p1.z *= inv; p1.w *= inv;
    float* dst = zout + (size_t)(rowBase + wave) * CODE_DEPTH + lane * 8;
    *(float4*)dst       = p0;
    *(float4*)(dst + 4) = p1;
}

// ---------------------------------------------------------------------------
// Kernel 2: scores = z @ cb16^T via f16 WMMA (16x16x32); fused argmin of
// cnorm[c] - 2*score, then gather codebook[idx] -> hard_codes.
// 512 threads = 16 waves share the same 16 rows; wave w sweeps code tiles
// w, w+16, ..., w+240. A fragments are hoisted into registers (the rows are
// invariant across the code sweep), so the hot loop is pure B-loads + WMMA.
// ---------------------------------------------------------------------------
__global__ __launch_bounds__(512)
void argmin_kernel(const float* __restrict__ z,
                   const _Float16* __restrict__ cb16,
                   const float* __restrict__ cnorm,
                   const float* __restrict__ codebook,
                   float* __restrict__ idxOut,
                   float* __restrict__ hcOut) {
    __shared__ _Float16 zs[16][256];    // 8 KB
    __shared__ float candB[16][16];
    __shared__ int   candI[16][16];
    __shared__ int   widx[16];

    const int tid     = threadIdx.x;
    const int wave    = tid >> 5;
    const int lane    = tid & 31;
    const int rowBase = blockIdx.x * 16;

    // stage the 16x256 z strip as f16
#pragma unroll
    for (int j = 0; j < 8; ++j) {
        const int e = tid * 8 + j;
        const int r = e >> 8, c = e & 255;
        zs[r][c] = (_Float16)z[(size_t)(rowBase + r) * CODE_DEPTH + c];
    }
    __syncthreads();

    const int m = lane & 15;
    const int h = lane >> 4;

    // hoist all 8 A fragments (16x32 f16 each) into registers: loads land
    // directly in the WMMA source register halves (no assembly movs).
    v16h afrag[8];
#pragma unroll
    for (int kc = 0; kc < 8; ++kc) {
        const int k0 = kc * 32 + 8 * h;
        *(v8h*)&afrag[kc]       = *(const v8h*)&zs[m][k0];
        *((v8h*)&afrag[kc] + 1) = *(const v8h*)&zs[m][k0 + 16];
    }

    float best[8];
    int   bidx[8];
#pragma unroll
    for (int v = 0; v < 8; ++v) { best[v] = 1e38f; bidx[v] = 0; }

    for (int t = wave; t < NUM_CODES / 16; t += 16) {
        const int code = t * 16 + m;
        const _Float16* brow = cb16 + (size_t)code * CODE_DEPTH;
        // cover L2 latency for the next code tile this wave will touch
        if (t + 16 < NUM_CODES / 16)
            __builtin_prefetch(brow + 256 * CODE_DEPTH, 0, 0);
        v8f acc = {};
#pragma unroll
        for (int kc = 0; kc < 8; ++kc) {
            const int k0 = kc * 32 + 8 * h;
            v16h b;
            *(v8h*)&b       = *(const v8h*)&brow[k0];
            *((v8h*)&b + 1) = *(const v8h*)&brow[k0 + 16];
            acc = __builtin_amdgcn_wmma_f32_16x16x32_f16(
                false, afrag[kc], false, b, (short)0, acc, false, false);
        }
        const float cn = cnorm[code];
#pragma unroll
        for (int v = 0; v < 8; ++v) {
            const float d = cn - 2.0f * acc[v];
            if (d < best[v]) { best[v] = d; bidx[v] = code; }
        }
    }

    // reduce across the 16 lanes of each half (rows v and v+8)
#pragma unroll
    for (int v = 0; v < 8; ++v) {
        float b = best[v]; int ix = bidx[v];
#pragma unroll
        for (int msk = 1; msk <= 8; msk <<= 1) {
            const float ob = __shfl_xor(b, msk, 32);
            const int   oi = __shfl_xor(ix, msk, 32);
            if (ob < b || (ob == b && oi < ix)) { b = ob; ix = oi; }
        }
        if (m == 0) { candB[wave][v + 8 * h] = b; candI[wave][v + 8 * h] = ix; }
    }
    __syncthreads();

    // reduce across waves (disjoint code subsets, same rows)
    if (tid < 16) {
        float b = candB[0][tid]; int ix = candI[0][tid];
        for (int w = 1; w < 16; ++w) {
            const float ob = candB[w][tid]; const int oi = candI[w][tid];
            if (ob < b || (ob == b && oi < ix)) { b = ob; ix = oi; }
        }
        widx[tid] = ix;
        idxOut[rowBase + tid] = (float)ix;
    }
    __syncthreads();

    // gather hard_codes = codebook[idx] (256 f32 per row, f32x4 copies)
    {
        const int r = tid >> 5;
        const float* src = codebook + (size_t)widx[r] * CODE_DEPTH;
        float* dst = hcOut + (size_t)(rowBase + r) * CODE_DEPTH;
        *(float4*)&dst[lane * 8]     = *(const float4*)&src[lane * 8];
        *(float4*)&dst[lane * 8 + 4] = *(const float4*)&src[lane * 8 + 4];
    }
}

// ---------------------------------------------------------------------------
extern "C" void kernel_launch(void* const* d_in, const int* in_sizes, int n_in,
                              void* d_out, int out_size, void* d_ws, size_t ws_size,
                              hipStream_t stream) {
    const float* x  = (const float*)d_in[0];
    const float* cb = (const float*)d_in[1];
    const float* W  = (const float*)d_in[2];
    const float* b  = (const float*)d_in[3];

    float* zout   = (float*)d_out;                               // 32768*256
    float* idxOut = zout + (size_t)N_TOKENS * CODE_DEPTH;        // 32768 (as f32)
    float* hcOut  = idxOut + N_TOKENS;                           // 32768*256

    float*    cnorm = (float*)d_ws;                              // 16 KB
    _Float16* cb16  = (_Float16*)((char*)d_ws + NUM_CODES * sizeof(float)); // 2 MB

    prep_kernel<<<NUM_CODES / 8, 256, 0, stream>>>(cb, cnorm, cb16);
    down_kernel<<<N_TOKENS / 16, 512, 0, stream>>>(x, W, b, zout);
    argmin_kernel<<<N_TOKENS / 16, 512, 0, stream>>>(zout, cb16, cnorm, cb,
                                                     idxOut, hcOut);
}
